// PointNet2_43894565765761
// MI455X (gfx1250) — matrix operand order repaired and checked by reading, not compiled, not run on hardware
//
#include <hip/hip_runtime.h>
#include <hip/hip_bf16.h>

typedef __attribute__((ext_vector_type(16))) _Float16 v16h;
typedef __attribute__((ext_vector_type(8)))  _Float16 v8h;
typedef __attribute__((ext_vector_type(8)))  float    v8f;

// ---------------------------------------------------------------------------
// WMMA helpers (CDNA5 wave32, V_WMMA_F32_16X16X32_F16)
// ---------------------------------------------------------------------------
__device__ __forceinline__ v8f wmma32(v16h a, v16h b, v8f c) {
  // 8 args: (neg_a, A, neg_b, B, c_mod, C, reuse_a, reuse_b)
  return __builtin_amdgcn_wmma_f32_16x16x32_f16(false, a, false, b, (short)0, c,
                                                false, false);
}

// A matrix 16x32 f16, row-major source [M][ld] (lanes 0-15: K 0..7 / 16..23,
// lanes 16-31: K 8..15 / 24..31 per ISA layout). ld, k0 multiples of 32 ->
// 16B-aligned vector loads (2 x ds_load_b128 / global_load_b128).
__device__ __forceinline__ v16h frag_a(const _Float16* base, int ld, int mt,
                                       int k0, int lane) {
  const _Float16* p =
      base + (size_t)(mt * 16 + (lane & 15)) * ld + k0 + ((lane & 16) ? 8 : 0);
  v8h lo = *(const v8h*)p;
  v8h hi = *(const v8h*)(p + 16);
  return __builtin_shufflevector(lo, hi, 0, 1, 2, 3, 4, 5, 6, 7, 8, 9, 10, 11,
                                 12, 13, 14, 15);
}

// B matrix 32x16 f16 from fragment-swizzled weights: tile (ks, nt) stores each
// lane's 16 K-halves contiguously (32B aligned) -> single v16h load.
__device__ __forceinline__ v16h frag_b(const _Float16* W, int NT, int nt,
                                       int ks, int lane) {
  return *(const v16h*)(W + ((size_t)(ks * NT + nt) * 32 + lane) * 16);
}

// ---------------------------------------------------------------------------
// CDNA5 async memory->LDS copy (no VGPR staging, tracked by ASYNCcnt)
// ---------------------------------------------------------------------------
__device__ __forceinline__ void async_copy_b128(unsigned lds_off,
                                                const void* gaddr) {
  asm volatile("global_load_async_to_lds_b128 %0, %1, off" ::"v"(lds_off),
               "v"((unsigned long long)(uintptr_t)gaddr)
               : "memory");
}
__device__ __forceinline__ void wait_async0() {
  asm volatile("s_wait_asynccnt 0x0" ::: "memory");
}

// ---------------------------------------------------------------------------
// Weight conversion: fp32 [K][N] -> fp16 fragment-swizzled, K padded to Kpad.
// dst layout: tile (ks=k/32, nt=n/16); within tile, lane = (k%32)/8%2*16 + n%16
// holds e = (k%32)/16*8 + k%8: dst[((ks*NT+nt)*32 + lane)*16 + e].
// ---------------------------------------------------------------------------
__global__ void cvt_pad_swz_kernel(const float* __restrict__ src,
                                   _Float16* __restrict__ dst, int K, int N,
                                   int Kpad) {
  int i = blockIdx.x * blockDim.x + threadIdx.x;
  if (i >= Kpad * N) return;
  int k = i / N, n = i - k * N;
  float v = (k < K) ? src[k * N + n] : 0.0f;
  int ks = k >> 5, kin = k & 31, khi = (kin >> 3) & 1, klo = ((kin >> 4) << 3) | (kin & 7);
  int nt = n >> 4, nn = n & 15;
  int NT = N >> 4;
  size_t d = ((size_t)((ks * NT + nt) * 32 + (khi * 16 + nn))) * 16 + klo;
  dst[d] = (_Float16)v;
}

// ---------------------------------------------------------------------------
// Farthest point sampling: 1 block per cloud, points cached in registers
// ---------------------------------------------------------------------------
template <int NPTS, int NSAMP>
__global__ void fps_kernel(const float* __restrict__ pos,
                           float* __restrict__ centers) {
  constexpr int TPB = 512;
  constexpr int PTS = NPTS / TPB;
  int b = blockIdx.x, tid = threadIdx.x;
  const float* P = pos + (size_t)b * NPTS * 3;

  float px[PTS], py[PTS], pz[PTS], mind[PTS];
#pragma unroll
  for (int i = 0; i < PTS; ++i) {
    int p = tid + i * TPB;
    px[i] = P[p * 3 + 0];
    py[i] = P[p * 3 + 1];
    pz[i] = P[p * 3 + 2];
    mind[i] = 3.0e38f;
  }

  __shared__ float s_val[TPB];
  __shared__ int s_idx[TPB];
  __shared__ float s_c[3];
  __shared__ int s_last;
  if (tid == 0) s_last = 0;
  __syncthreads();

  for (int s = 0; s < NSAMP; ++s) {
    if (tid == 0) {
      int last = s_last;
      float cx = P[last * 3 + 0], cy = P[last * 3 + 1], cz = P[last * 3 + 2];
      s_c[0] = cx; s_c[1] = cy; s_c[2] = cz;
      float* C = centers + ((size_t)b * NSAMP + s) * 3;
      C[0] = cx; C[1] = cy; C[2] = cz;
    }
    __syncthreads();
    float cx = s_c[0], cy = s_c[1], cz = s_c[2];
    float best = -1.0f;
    int bi = 0;
#pragma unroll
    for (int i = 0; i < PTS; ++i) {
      float dx = px[i] - cx, dy = py[i] - cy, dz = pz[i] - cz;
      float d = dx * dx + dy * dy + dz * dz;
      mind[i] = fminf(mind[i], d);
      if (mind[i] > best) { best = mind[i]; bi = tid + i * TPB; }
    }
    s_val[tid] = best;
    s_idx[tid] = bi;
    __syncthreads();
    for (int off = TPB / 2; off > 0; off >>= 1) {
      if (tid < off && s_val[tid + off] > s_val[tid]) {
        s_val[tid] = s_val[tid + off];
        s_idx[tid] = s_idx[tid + off];
      }
      __syncthreads();
    }
    if (tid == 0) s_last = s_idx[0];
    __syncthreads();
  }
}

// ---------------------------------------------------------------------------
// Deterministic ball query: striped scan + LDS prefix sum, first K in ball
// (index order), pad with first neighbor (center itself always qualifies).
// ---------------------------------------------------------------------------
__global__ void ballquery_kernel(const float* __restrict__ pos,
                                 const float* __restrict__ centers, int NPTS,
                                 int S, float r2, int* __restrict__ nbr,
                                 int* __restrict__ cnt) {
  constexpr int K = 64;
  int g = blockIdx.x;
  int b = g / S;
  int tid = threadIdx.x;  // 64 threads
  const float* P = pos + (size_t)b * NPTS * 3;
  float cx = centers[(size_t)g * 3 + 0];
  float cy = centers[(size_t)g * 3 + 1];
  float cz = centers[(size_t)g * 3 + 2];

  int stripe = NPTS / 64;
  int lo = tid * stripe, hi = lo + stripe;
  int c_local = 0;
  for (int i = lo; i < hi; ++i) {
    float dx = P[i * 3 + 0] - cx, dy = P[i * 3 + 1] - cy,
          dz = P[i * 3 + 2] - cz;
    if (dx * dx + dy * dy + dz * dz <= r2) c_local++;
  }
  __shared__ int s_off[65];
  __shared__ int s_cnt[64];
  s_cnt[tid] = c_local;
  __syncthreads();
  if (tid == 0) {
    int acc = 0;
    for (int t = 0; t < 64; ++t) { s_off[t] = acc; acc += s_cnt[t]; }
    s_off[64] = acc;
  }
  __syncthreads();
  int off = s_off[tid];
  int total = s_off[64];
  int* NB = nbr + (size_t)g * K;
  for (int i = lo; i < hi && off < K; ++i) {
    float dx = P[i * 3 + 0] - cx, dy = P[i * 3 + 1] - cy,
          dz = P[i * 3 + 2] - cz;
    if (dx * dx + dy * dy + dz * dz <= r2) { NB[off] = i; off++; }
  }
  __syncthreads();
  int c = total < K ? total : K;
  if (tid == 0) cnt[g] = c;
  int first = NB[0];
  for (int j = c + tid; j < K; j += 64) NB[j] = first;
}

// ---------------------------------------------------------------------------
// SA layer: gather group features -> 2-layer MLP via WMMA -> masked max-pool
// One 4-wave block per (batch, center) group of 64 neighbors. Weights staged
// in LDS via CDNA5 async DMA, overlapped with the feature gather.
// ---------------------------------------------------------------------------
template <int CIN, int CIN_PAD, int CMID, int COUT>
__global__ void sa_mlp_kernel(
    const float* __restrict__ feat_in, const float* __restrict__ pos_in,
    const float* __restrict__ centers, const int* __restrict__ nbr,
    const int* __restrict__ cnt, const _Float16* __restrict__ W1,
    const float* __restrict__ b1, const _Float16* __restrict__ W2,
    const float* __restrict__ b2, float* __restrict__ out, int NPTS, int S) {
  constexpr int CF = CIN - 3;
  constexpr int K = 64;
  extern __shared__ char dynsmem[];
  _Float16* sW1 = (_Float16*)dynsmem;           // CIN_PAD x CMID (swizzled)
  _Float16* sW2 = sW1 + CIN_PAD * CMID;         // CMID x COUT (swizzled)
  _Float16* sF  = sW2 + CMID * COUT;            // K x CIN_PAD
  _Float16* sH  = sF + K * CIN_PAD;             // K x CMID
  __shared__ int s_nbr[K];
  __shared__ float s_ctr[3];

  int g = blockIdx.x;
  int b = g / S;
  int tid = threadIdx.x, lane = tid & 31, wv = tid >> 5;

  // async DMA of both weight blocks into LDS (16B per lane per op;
  // trip counts uniform across the wave -> EXEC all ones)
  constexpr int NC1 = CIN_PAD * CMID / 8;
  for (int i = tid; i < NC1; i += 128)
    async_copy_b128((unsigned)(uintptr_t)(sW1 + i * 8), W1 + i * 8);
  constexpr int NC2 = CMID * COUT / 8;
  for (int i = tid; i < NC2; i += 128)
    async_copy_b128((unsigned)(uintptr_t)(sW2 + i * 8), W2 + i * 8);

  if (tid < K) s_nbr[tid] = nbr[(size_t)g * K + tid];
  if (tid < 3) s_ctr[tid] = centers[(size_t)g * 3 + tid];
  __syncthreads();

  // gather features: [x[nbr], pos[nbr]-center, 0-pad] (overlaps weight DMA)
  for (int e = tid; e < K * CIN_PAD; e += blockDim.x) {
    int j = e / CIN_PAD, k = e - j * CIN_PAD;
    int p = s_nbr[j];
    float v = 0.0f;
    if (k < CF) {
      v = feat_in[((size_t)b * NPTS + p) * CF + k];
    } else if (k < CF + 3) {
      int d = k - CF;
      v = pos_in[((size_t)b * NPTS + p) * 3 + d] - s_ctr[d];
    }
    sF[e] = (_Float16)v;
  }
  wait_async0();
  __syncthreads();

  // layer 1: [64 x CIN_PAD] x [CIN_PAD x CMID] -> relu -> sH (f16)
  for (int nt = wv; nt < CMID / 16; nt += 4) {
    int n = nt * 16 + (lane & 15);
    float bias = b1[n];
    for (int mt = 0; mt < 4; ++mt) {
      v8f acc = {};
#pragma unroll
      for (int ks = 0; ks < CIN_PAD / 32; ++ks)
        acc = wmma32(frag_a(sF, CIN_PAD, mt, ks * 32, lane),
                     frag_b(sW1, CMID / 16, nt, ks, lane), acc);
      int mb = mt * 16 + ((lane & 16) ? 8 : 0);
#pragma unroll
      for (int r = 0; r < 8; ++r) {
        float v = acc[r] + bias;
        v = v > 0.0f ? v : 0.0f;
        sH[(mb + r) * CMID + n] = (_Float16)v;
      }
    }
  }
  __syncthreads();

  // layer 2 + masked max-pool over neighbors
  int c = cnt[g];
  const float NEG = -3.0e38f;
  for (int nt = wv; nt < COUT / 16; nt += 4) {
    int n = nt * 16 + (lane & 15);
    float bias = b2[n];
    float colmax = NEG;
    for (int mt = 0; mt < 4; ++mt) {
      v8f acc = {};
#pragma unroll
      for (int ks = 0; ks < CMID / 32; ++ks)
        acc = wmma32(frag_a(sH, CMID, mt, ks * 32, lane),
                     frag_b(sW2, COUT / 16, nt, ks, lane), acc);
      int mb = mt * 16 + ((lane & 16) ? 8 : 0);
#pragma unroll
      for (int r = 0; r < 8; ++r) {
        float v = acc[r] + bias;
        if (mb + r >= c) v = NEG;
        colmax = fmaxf(colmax, v);
      }
    }
    colmax = fmaxf(colmax, __shfl_xor(colmax, 16, 32));
    if (lane < 16) out[(size_t)g * COUT + n] = colmax;
  }
}

// ---------------------------------------------------------------------------
// Global SA: MLP 288(pad of 259) -> 256 -> 512 -> 1024 on 64-row tiles.
// B fragments read straight from L2-resident swizzled weights (1KB/wave/frag).
// ---------------------------------------------------------------------------
__global__ void global_mlp_kernel(
    const float* __restrict__ x2, const float* __restrict__ p2,
    const _Float16* __restrict__ W1, const float* __restrict__ b1,
    const _Float16* __restrict__ W2, const float* __restrict__ b2,
    const _Float16* __restrict__ W3, const float* __restrict__ b3,
    float* __restrict__ g3) {
  const int CIN = 256, CINP = 288, C1 = 256, C2 = 512, C3 = 1024;
  extern __shared__ char dynsmem[];
  _Float16* sF  = (_Float16*)dynsmem;  // 64 x 288
  _Float16* sH1 = sF + 64 * CINP;      // 64 x 256
  _Float16* sH2 = sH1 + 64 * C1;       // 64 x 512
  int rb = blockIdx.x;
  int tid = threadIdx.x, lane = tid & 31, wv = tid >> 5;

  for (int e = tid; e < 64 * CINP; e += blockDim.x) {
    int r = e / CINP, k = e - r * CINP;
    size_t grow = (size_t)rb * 64 + r;
    float v = 0.0f;
    if (k < CIN) v = x2[grow * CIN + k];
    else if (k < CIN + 3) v = p2[grow * 3 + (k - CIN)];
    sF[e] = (_Float16)v;
  }
  // warm L2 with layer-3 weights while layer 1/2 run
  for (size_t e = tid; e < (size_t)C2 * C3; e += (size_t)blockDim.x * 512)
    __builtin_prefetch(&W3[e], 0, 1);
  __syncthreads();

  for (int nt = wv; nt < C1 / 16; nt += 4) {
    int n = nt * 16 + (lane & 15);
    float bias = b1[n];
    for (int mt = 0; mt < 4; ++mt) {
      v8f acc = {};
#pragma unroll
      for (int ks = 0; ks < CINP / 32; ++ks)
        acc = wmma32(frag_a(sF, CINP, mt, ks * 32, lane),
                     frag_b(W1, C1 / 16, nt, ks, lane), acc);
      int mb = mt * 16 + ((lane & 16) ? 8 : 0);
#pragma unroll
      for (int r = 0; r < 8; ++r) {
        float v = acc[r] + bias;
        v = v > 0.0f ? v : 0.0f;
        sH1[(mb + r) * C1 + n] = (_Float16)v;
      }
    }
  }
  __syncthreads();

  for (int nt = wv; nt < C2 / 16; nt += 4) {
    int n = nt * 16 + (lane & 15);
    float bias = b2[n];
    for (int mt = 0; mt < 4; ++mt) {
      v8f acc = {};
#pragma unroll
      for (int ks = 0; ks < C1 / 32; ++ks)
        acc = wmma32(frag_a(sH1, C1, mt, ks * 32, lane),
                     frag_b(W2, C2 / 16, nt, ks, lane), acc);
      int mb = mt * 16 + ((lane & 16) ? 8 : 0);
#pragma unroll
      for (int r = 0; r < 8; ++r) {
        float v = acc[r] + bias;
        v = v > 0.0f ? v : 0.0f;
        sH2[(mb + r) * C2 + n] = (_Float16)v;
      }
    }
  }
  __syncthreads();

  for (int nt = wv; nt < C3 / 16; nt += 4) {
    int n = nt * 16 + (lane & 15);
    float bias = b3[n];
    for (int mt = 0; mt < 4; ++mt) {
      v8f acc = {};
#pragma unroll
      for (int ks = 0; ks < C2 / 32; ++ks)
        acc = wmma32(frag_a(sH2, C2, mt, ks * 32, lane),
                     frag_b(W3, C3 / 16, nt, ks, lane), acc);
      int mb = mt * 16 + ((lane & 16) ? 8 : 0);
      size_t rowbase = (size_t)rb * 64;
#pragma unroll
      for (int r = 0; r < 8; ++r)
        g3[(rowbase + mb + r) * C3 + n] = acc[r] + bias;
    }
  }
}

// ---------------------------------------------------------------------------
// Global max pool over 512 points per batch
// ---------------------------------------------------------------------------
__global__ void maxpool_rows_kernel(const float* __restrict__ g3,
                                    float* __restrict__ gmax) {
  int b = blockIdx.x;
  int j = blockIdx.y * blockDim.x + threadIdx.x;  // 0..1023
  float m = -3.0e38f;
  const float* src = g3 + (size_t)b * 512 * 1024 + j;
  for (int s = 0; s < 512; ++s) m = fmaxf(m, src[(size_t)s * 1024]);
  gmax[b * 1024 + j] = m;
}

// ---------------------------------------------------------------------------
// Head dense layers (8 rows only — plain fp32)
// ---------------------------------------------------------------------------
__global__ void dense_kernel(const float* __restrict__ in,
                             const float* __restrict__ W,
                             const float* __restrict__ bias,
                             float* __restrict__ out, int Kdim, int Ndim,
                             int relu) {
  int b = blockIdx.x;
  int j = blockIdx.y * blockDim.x + threadIdx.x;
  if (j >= Ndim) return;
  float acc = bias[j];
  for (int k = 0; k < Kdim; ++k) acc += in[b * Kdim + k] * W[k * Ndim + j];
  if (relu) acc = fmaxf(acc, 0.0f);
  out[b * Ndim + j] = acc;
}

// ---------------------------------------------------------------------------
// Host launcher
// ---------------------------------------------------------------------------
extern "C" void kernel_launch(void* const* d_in, const int* in_sizes, int n_in,
                              void* d_out, int out_size, void* d_ws,
                              size_t ws_size, hipStream_t stream) {
  (void)in_sizes; (void)n_in; (void)out_size; (void)ws_size;
  const float* x      = (const float*)d_in[0];
  const float* pos    = (const float*)d_in[1];
  const float* sa1_w1 = (const float*)d_in[2];
  const float* sa1_b1 = (const float*)d_in[3];
  const float* sa1_w2 = (const float*)d_in[4];
  const float* sa1_b2 = (const float*)d_in[5];
  const float* sa2_w1 = (const float*)d_in[6];
  const float* sa2_b1 = (const float*)d_in[7];
  const float* sa2_w2 = (const float*)d_in[8];
  const float* sa2_b2 = (const float*)d_in[9];
  const float* sa3_w1 = (const float*)d_in[10];
  const float* sa3_b1 = (const float*)d_in[11];
  const float* sa3_w2 = (const float*)d_in[12];
  const float* sa3_b2 = (const float*)d_in[13];
  const float* sa3_w3 = (const float*)d_in[14];
  const float* sa3_b3 = (const float*)d_in[15];
  const float* hd_w1  = (const float*)d_in[16];
  const float* hd_b1  = (const float*)d_in[17];
  const float* hd_w2  = (const float*)d_in[18];
  const float* hd_b2  = (const float*)d_in[19];
  const float* hd_w3  = (const float*)d_in[20];
  const float* hd_b3  = (const float*)d_in[21];

  char* ws = (char*)d_ws;
  size_t off = 0;
  auto take = [&](size_t bytes) -> char* {
    char* p = ws + off;
    off += (bytes + 255) & ~(size_t)255;
    return p;
  };
  _Float16* W1A = (_Float16*)take((size_t)32 * 64 * 2);
  _Float16* W1B = (_Float16*)take((size_t)64 * 128 * 2);
  _Float16* W2A = (_Float16*)take((size_t)160 * 128 * 2);
  _Float16* W2B = (_Float16*)take((size_t)128 * 256 * 2);
  _Float16* W3A = (_Float16*)take((size_t)288 * 256 * 2);
  _Float16* W3B = (_Float16*)take((size_t)256 * 512 * 2);
  _Float16* W3C = (_Float16*)take((size_t)512 * 1024 * 2);
  float* p1  = (float*)take((size_t)8 * 2048 * 3 * 4);
  int*   cnt1 = (int*)take((size_t)8 * 2048 * 4);
  int*   nbr1 = (int*)take((size_t)8 * 2048 * 64 * 4);
  float* x1  = (float*)take((size_t)8 * 2048 * 128 * 4);
  float* p2  = (float*)take((size_t)8 * 512 * 3 * 4);
  int*   cnt2 = (int*)take((size_t)8 * 512 * 4);
  int*   nbr2 = (int*)take((size_t)8 * 512 * 64 * 4);
  float* x2  = (float*)take((size_t)8 * 512 * 256 * 4);
  float* g3  = (float*)take((size_t)4096 * 1024 * 4);
  float* gmax = (float*)take((size_t)8 * 1024 * 4);
  float* h1  = (float*)take((size_t)8 * 512 * 4);
  float* h2  = (float*)take((size_t)8 * 256 * 4);

  auto cvt = [&](const float* src, _Float16* dst, int K, int N, int Kpad) {
    int n = Kpad * N;
    cvt_pad_swz_kernel<<<(n + 255) / 256, 256, 0, stream>>>(src, dst, K, N,
                                                            Kpad);
  };
  cvt(sa1_w1, W1A, 6, 64, 32);
  cvt(sa1_w2, W1B, 64, 128, 64);
  cvt(sa2_w1, W2A, 131, 128, 160);
  cvt(sa2_w2, W2B, 128, 256, 128);
  cvt(sa3_w1, W3A, 259, 256, 288);
  cvt(sa3_w2, W3B, 256, 512, 256);
  cvt(sa3_w3, W3C, 512, 1024, 512);

  // SA1
  fps_kernel<4096, 2048><<<8, 512, 0, stream>>>(pos, p1);
  ballquery_kernel<<<8 * 2048, 64, 0, stream>>>(pos, p1, 4096, 2048,
                                                0.2f * 0.2f, nbr1, cnt1);
  {
    size_t sh = (size_t)(32 * 64 + 64 * 128 + 64 * 32 + 64 * 64) * 2;
    sa_mlp_kernel<6, 32, 64, 128><<<8 * 2048, 128, sh, stream>>>(
        x, pos, p1, nbr1, cnt1, W1A, sa1_b1, W1B, sa1_b2, x1, 4096, 2048);
  }
  // SA2
  fps_kernel<2048, 512><<<8, 512, 0, stream>>>(p1, p2);
  ballquery_kernel<<<8 * 512, 64, 0, stream>>>(p1, p2, 2048, 512, 0.4f * 0.4f,
                                               nbr2, cnt2);
  {
    size_t sh = (size_t)(160 * 128 + 128 * 256 + 64 * 160 + 64 * 128) * 2;
    sa_mlp_kernel<131, 160, 128, 256><<<8 * 512, 128, sh, stream>>>(
        x1, p1, p2, nbr2, cnt2, W2A, sa2_b1, W2B, sa2_b2, x2, 2048, 512);
  }
  // Global SA
  {
    size_t sh = (size_t)(64 * 288 + 64 * 256 + 64 * 512) * 2;
    global_mlp_kernel<<<64, 128, sh, stream>>>(x2, p2, W3A, sa3_b1, W3B,
                                               sa3_b2, W3C, sa3_b3, g3);
  }
  maxpool_rows_kernel<<<dim3(8, 4), 256, 0, stream>>>(g3, gmax);
  // Head
  dense_kernel<<<dim3(8, 2), 256, 0, stream>>>(gmax, hd_w1, hd_b1, h1, 1024,
                                               512, 1);
  dense_kernel<<<dim3(8, 1), 256, 0, stream>>>(h1, hd_w2, hd_b2, h2, 512, 256,
                                               1);
  dense_kernel<<<dim3(8, 1), 256, 0, stream>>>(h2, hd_w3, hd_b3,
                                               (float*)d_out, 256, 40, 0);
}
